// PointnetFPModule_16260746183081
// MI455X (gfx1250) — matrix operand order repaired and checked by reading, not compiled, not run on hardware
//
#include <hip/hip_runtime.h>
#include <hip/hip_bf16.h>

// ---------------------------------------------------------------------------
// Problem constants (from reference setup_inputs)
// ---------------------------------------------------------------------------
#define B_SZ   8
#define N_PTS  4096
#define M_PTS  1024
#define C1_DIM 256
#define C2_DIM 512
#define M_ROWS (B_SZ * N_PTS)      // 32768 rows in the fused GEMMs
#define K1_DIM (C1_DIM + C2_DIM)   // 768
#define BN_EPS 1e-3f

// GEMM tiling
#define KC        128              // K-chunk staged in LDS (divides 768 and 512)
#define SB_STRIDE 136              // KC + 8 pad (ushorts) -> bank-conflict-free ds_load_b128

// Workspace layout (all offsets 256-byte aligned)
#define OFF_IDX  ((size_t)0)                                   // 32768*3 int
#define OFF_WGT  (OFF_IDX + (size_t)M_ROWS * 3 * 4)            // 32768*3 f32
#define OFF_WT1  (OFF_WGT + (size_t)M_ROWS * 3 * 4)            // 512*768 bf16
#define OFF_WT2  (OFF_WT1 + (size_t)512 * 768 * 2)             // 512*512 bf16
#define OFF_WT3  (OFF_WT2 + (size_t)512 * 512 * 2)             // 256*512 bf16
#define OFF_X    (OFF_WT3 + (size_t)256 * 512 * 2)             // 32768*768 bf16
#define OFF_H1   (OFF_X   + (size_t)M_ROWS * K1_DIM * 2)       // 32768*512 bf16
#define OFF_H2   (OFF_H1  + (size_t)M_ROWS * 512 * 2)          // 32768*512 bf16

typedef __attribute__((ext_vector_type(16))) __bf16 v16bf;
typedef __attribute__((ext_vector_type(8)))  float  v8f;
typedef int v4i_gcc __attribute__((vector_size(16)));   // matches builtin param type

union Frag16 { v16bf v; uint4 q[2]; };

__device__ __forceinline__ unsigned short f32_to_bf16(float f) {
    unsigned int u = __float_as_uint(f);
    unsigned int r = 0x7FFFu + ((u >> 16) & 1u);   // round-to-nearest-even
    return (unsigned short)((u + r) >> 16);
}

// ---------------------------------------------------------------------------
// gfx1250 async global->LDS copy (ASYNCcnt path), with portable fallbacks.
// Builtin signature (from hipcc diagnostic): param0 = int4 __device__* (AS1),
// param1 = LDS pointer (AS3), then offset, cpol.
// ---------------------------------------------------------------------------
__device__ __forceinline__ void async_copy_b128(const unsigned short* gsrc,
                                                unsigned short* lds_dst) {
#if defined(__has_builtin) && __has_builtin(__builtin_amdgcn_global_load_async_to_lds_b128)
    __builtin_amdgcn_global_load_async_to_lds_b128(
        (__attribute__((address_space(1))) v4i_gcc*)gsrc,
        (__attribute__((address_space(3))) v4i_gcc*)lds_dst, 0, 0);
#else
    *(uint4*)lds_dst = *(const uint4*)gsrc;        // synchronous fallback
#endif
}

__device__ __forceinline__ void wait_async_all() {
#if defined(__has_builtin) && __has_builtin(__builtin_amdgcn_s_wait_asynccnt)
    __builtin_amdgcn_s_wait_asynccnt(0);
#else
    asm volatile("s_wait_asynccnt 0" ::: "memory");
#endif
}

// ---------------------------------------------------------------------------
// Repack W (K x N, f32 row-major) -> W^T (N x K, bf16 row-major) so the WMMA
// B-fragment (K-contiguous per output column) is two contiguous 16B loads.
// ---------------------------------------------------------------------------
__global__ __launch_bounds__(256)
void repack_w_bf16(const float* __restrict__ W, unsigned short* __restrict__ WT,
                   int K, int N) {
    int id = blockIdx.x * 256 + threadIdx.x;
    if (id >= K * N) return;
    int k = id / N, n = id % N;
    WT[(size_t)n * K + k] = f32_to_bf16(W[id]);
}

// ---------------------------------------------------------------------------
// three_nn: one thread per query point; known xyz staged in LDS (12 KB/batch).
// ---------------------------------------------------------------------------
__global__ __launch_bounds__(256)
void three_nn_kernel(const float* __restrict__ unknown, const float* __restrict__ known,
                     int* __restrict__ idxOut, float* __restrict__ wOut) {
    __shared__ float kx[M_PTS], ky[M_PTS], kz[M_PTS];
    const int blocksPerBatch = N_PTS / 256;            // 16
    const int b = blockIdx.x / blocksPerBatch;
    const int q = (blockIdx.x % blocksPerBatch) * 256 + threadIdx.x;

    for (int j = threadIdx.x; j < M_PTS; j += 256) {
        const float* kp = known + ((size_t)b * M_PTS + j) * 3;
        kx[j] = kp[0]; ky[j] = kp[1]; kz[j] = kp[2];
    }
    __syncthreads();

    const float* up = unknown + ((size_t)b * N_PTS + q) * 3;
    const float ux = up[0], uy = up[1], uz = up[2];

    float d0 = 1e30f, d1 = 1e30f, d2 = 1e30f;
    int   i0 = 0,     i1 = 0,     i2 = 0;
    for (int j = 0; j < M_PTS; ++j) {
        float dx = ux - kx[j], dy = uy - ky[j], dz = uz - kz[j];
        float d  = dx * dx + dy * dy + dz * dz;
        if (d < d0)      { d2 = d1; i2 = i1; d1 = d0; i1 = i0; d0 = d; i0 = j; }
        else if (d < d1) { d2 = d1; i2 = i1; d1 = d;  i1 = j; }
        else if (d < d2) { d2 = d;  i2 = j; }
    }
    float r0 = 1.0f / (d0 + 1e-8f);
    float r1 = 1.0f / (d1 + 1e-8f);
    float r2 = 1.0f / (d2 + 1e-8f);
    float inv = 1.0f / (r0 + r1 + r2);

    size_t o = ((size_t)b * N_PTS + q) * 3;
    idxOut[o + 0] = i0; idxOut[o + 1] = i1; idxOut[o + 2] = i2;
    wOut[o + 0] = r0 * inv; wOut[o + 1] = r1 * inv; wOut[o + 2] = r2 * inv;
}

// ---------------------------------------------------------------------------
// Weighted 3-gather of known_feats + concat unknow_feats -> fused bf16 X row.
// ---------------------------------------------------------------------------
__global__ __launch_bounds__(256)
void interp_concat_kernel(const float* __restrict__ known_feats,
                          const float* __restrict__ unknow_feats,
                          const int* __restrict__ idx, const float* __restrict__ w,
                          unsigned short* __restrict__ X) {
    const int row = blockIdx.x;            // 0 .. M_ROWS-1
    const int b   = row / N_PTS;
    const int t   = threadIdx.x;

    size_t o = (size_t)row * 3;
    const int   i0 = idx[o + 0], i1 = idx[o + 1], i2 = idx[o + 2];
    const float w0 = w[o + 0],   w1 = w[o + 1],   w2 = w[o + 2];

    const float* f0 = known_feats + ((size_t)b * M_PTS + i0) * C2_DIM;
    const float* f1 = known_feats + ((size_t)b * M_PTS + i1) * C2_DIM;
    const float* f2 = known_feats + ((size_t)b * M_PTS + i2) * C2_DIM;
    unsigned short* xr = X + (size_t)row * K1_DIM;

    #pragma unroll
    for (int c = t; c < C2_DIM; c += 256)
        xr[c] = f32_to_bf16(w0 * f0[c] + w1 * f1[c] + w2 * f2[c]);

    const float* uf = unknow_feats + (size_t)row * C1_DIM;
    for (int c = t; c < C1_DIM; c += 256)
        xr[C2_DIM + c] = f32_to_bf16(uf[c]);
}

// ---------------------------------------------------------------------------
// Fused GEMM + BatchNorm + ReLU, v_wmma_f32_16x16x32_bf16.
//   X  : M x K bf16 row-major (activations, streamed from global)
//   WT : N x K bf16 row-major (weights, async-staged into LDS per 64-col strip)
// Block = 8 waves sharing one 64-wide N strip; each wave owns a 32x64 output
// tile (2 A-frags x 4 B-frags, 8 accumulators -> 8 WMMA / 12 b128-loads per
// 32-wide k-step). B panel double-buffered in LDS via async-to-LDS copies.
// ---------------------------------------------------------------------------
__global__ __launch_bounds__(256)
void gemm_bn_relu(const unsigned short* __restrict__ X,
                  const unsigned short* __restrict__ WT,
                  const float* __restrict__ gg, const float* __restrict__ bb,
                  const float* __restrict__ mm, const float* __restrict__ vv,
                  unsigned short* __restrict__ outBf, float* __restrict__ outF,
                  int N, int K) {
    __shared__ __align__(16) unsigned short sB[2][64 * SB_STRIDE];

    const int t       = threadIdx.x;
    const int lane    = t & 31;
    const int wave    = t >> 5;
    const int nTiles  = N >> 6;                     // 64-wide N strips
    const int nTile   = blockIdx.x % nTiles;
    const int mSuper  = blockIdx.x / nTiles;        // 256-row super-tile
    const int rowBase = mSuper * 256 + wave * 32;
    const int colBase = nTile << 6;
    const int hi      = lane >> 4;                  // K-half selector
    const int lane16  = lane & 15;

    // A fragment pointers: rows rowBase+lane16 and rowBase+16+lane16.
    // Elements 0..7 = K[kb..kb+7], 8..15 = K[kb+16..kb+23], kb = 8*hi.
    const unsigned short* aPtr0 = X + (size_t)(rowBase + lane16)      * K + hi * 8;
    const unsigned short* aPtr1 = X + (size_t)(rowBase + 16 + lane16) * K + hi * 8;

    // B copy mapping: 64 cols x (KC/8)=16 segments of 16B -> 1024 segs, 4/thread.
    const int nChunks = K / KC;

    v8f acc[2][4];
    #pragma unroll
    for (int mi = 0; mi < 2; ++mi)
        #pragma unroll
        for (int j = 0; j < 4; ++j) acc[mi][j] = (v8f){};

    // Issue chunk 0
    {
        #pragma unroll
        for (int i = 0; i < 4; ++i) {
            int s  = t + i * 256;
            int c  = s >> 4;
            int ko = (s & 15) * 8;
            async_copy_b128(WT + (size_t)(colBase + c) * K + ko,
                            &sB[0][c * SB_STRIDE + ko]);
        }
    }

    for (int ch = 0; ch < nChunks; ++ch) {
        const int k0  = ch * KC;
        const int buf = ch & 1;

        wait_async_all();           // chunk `ch` copies complete (in-order cnt)
        __syncthreads();            // visible to all waves; prev compute done

        if (ch + 1 < nChunks) {     // overlap next chunk's copy with compute
            const int kn = k0 + KC;
            #pragma unroll
            for (int i = 0; i < 4; ++i) {
                int s  = t + i * 256;
                int c  = s >> 4;
                int ko = (s & 15) * 8;
                async_copy_b128(WT + (size_t)(colBase + c) * K + kn + ko,
                                &sB[buf ^ 1][c * SB_STRIDE + ko]);
            }
        }

        #pragma unroll
        for (int kk = 0; kk < KC; kk += 32) {
            Frag16 a0, a1;
            a0.q[0] = *(const uint4*)(aPtr0 + k0 + kk);
            a0.q[1] = *(const uint4*)(aPtr0 + k0 + kk + 16);
            a1.q[0] = *(const uint4*)(aPtr1 + k0 + kk);
            a1.q[1] = *(const uint4*)(aPtr1 + k0 + kk + 16);

            #pragma unroll
            for (int j = 0; j < 4; ++j) {
                // B fragment: col = j*16+lane16, elements 0..7 = K[kk+8*hi..+7], 8..15 = K[kk+16+8*hi..+7]
                const unsigned short* bl =
                    &sB[buf][(j * 16 + lane16) * SB_STRIDE + kk + 8 * hi];
                Frag16 b;
                b.q[0] = *(const uint4*)(bl);
                b.q[1] = *(const uint4*)(bl + 16);
                acc[0][j] = __builtin_amdgcn_wmma_f32_16x16x32_bf16(
                    false, a0.v, false, b.v, (short)0, acc[0][j], false, false);
                acc[1][j] = __builtin_amdgcn_wmma_f32_16x16x32_bf16(
                    false, a1.v, false, b.v, (short)0, acc[1][j], false, false);
            }
        }
        __syncthreads();            // all reads of sB[buf] done before overwrite
    }

    // Fused BatchNorm(+ReLU) epilogue: per-column scale/bias, per-lane constant.
    #pragma unroll
    for (int j = 0; j < 4; ++j) {
        const int col = colBase + j * 16 + lane16;
        const float s = gg[col] * rsqrtf(vv[col] + BN_EPS);
        const float tt = bb[col] - mm[col] * s;
        #pragma unroll
        for (int mi = 0; mi < 2; ++mi) {
            #pragma unroll
            for (int r = 0; r < 8; ++r) {
                const int row = rowBase + mi * 16 + r + 8 * hi;
                const float val = fmaxf(acc[mi][j][r] * s + tt, 0.0f);
                if (outBf) outBf[(size_t)row * N + col] = f32_to_bf16(val);
                else       outF [(size_t)row * N + col] = val;
            }
        }
    }
}

// ---------------------------------------------------------------------------
// Launcher
// ---------------------------------------------------------------------------
extern "C" void kernel_launch(void* const* d_in, const int* in_sizes, int n_in,
                              void* d_out, int out_size, void* d_ws, size_t ws_size,
                              hipStream_t stream) {
    const float* unknown      = (const float*)d_in[0];
    const float* known        = (const float*)d_in[1];
    const float* unknow_feats = (const float*)d_in[2];
    const float* known_feats  = (const float*)d_in[3];
    const float* W1  = (const float*)d_in[4];
    const float* g1  = (const float*)d_in[5];
    const float* b1  = (const float*)d_in[6];
    const float* mu1 = (const float*)d_in[7];
    const float* v1  = (const float*)d_in[8];
    const float* W2  = (const float*)d_in[9];
    const float* g2  = (const float*)d_in[10];
    const float* b2  = (const float*)d_in[11];
    const float* mu2 = (const float*)d_in[12];
    const float* v2  = (const float*)d_in[13];
    const float* W3  = (const float*)d_in[14];
    const float* g3  = (const float*)d_in[15];
    const float* b3  = (const float*)d_in[16];
    const float* mu3 = (const float*)d_in[17];
    const float* v3  = (const float*)d_in[18];

    char* ws = (char*)d_ws;
    int*            idx = (int*)           (ws + OFF_IDX);
    float*          wgt = (float*)         (ws + OFF_WGT);
    unsigned short* WT1 = (unsigned short*)(ws + OFF_WT1);
    unsigned short* WT2 = (unsigned short*)(ws + OFF_WT2);
    unsigned short* WT3 = (unsigned short*)(ws + OFF_WT3);
    unsigned short* X   = (unsigned short*)(ws + OFF_X);
    unsigned short* H1  = (unsigned short*)(ws + OFF_H1);
    unsigned short* H2  = (unsigned short*)(ws + OFF_H2);
    float*          OUT = (float*)d_out;

    // 1) Repack weights to bf16 W^T
    repack_w_bf16<<<(K1_DIM * 512 + 255) / 256, 256, 0, stream>>>(W1, WT1, K1_DIM, 512);
    repack_w_bf16<<<(512 * 512 + 255) / 256, 256, 0, stream>>>(W2, WT2, 512, 512);
    repack_w_bf16<<<(512 * 256 + 255) / 256, 256, 0, stream>>>(W3, WT3, 512, 256);

    // 2) three_nn
    three_nn_kernel<<<B_SZ * (N_PTS / 256), 256, 0, stream>>>(unknown, known, idx, wgt);

    // 3) interpolate + concat + bf16 convert into fused X
    interp_concat_kernel<<<M_ROWS, 256, 0, stream>>>(known_feats, unknow_feats, idx, wgt, X);

    // 4) Three fused WMMA GEMM + BN + ReLU layers
    //    blocks = (M/256) * (N/64)
    const int blocks1 = (M_ROWS / 256) * (512 / 64);   // 1024
    gemm_bn_relu<<<blocks1, 256, 0, stream>>>(X, WT1, g1, b1, mu1, v1,
                                              H1, nullptr, 512, K1_DIM);
    const int blocks2 = (M_ROWS / 256) * (512 / 64);   // 1024
    gemm_bn_relu<<<blocks2, 256, 0, stream>>>(H1, WT2, g2, b2, mu2, v2,
                                              H2, nullptr, 512, 512);
    const int blocks3 = (M_ROWS / 256) * (256 / 64);   // 512
    gemm_bn_relu<<<blocks3, 256, 0, stream>>>(H2, WT3, g3, b3, mu3, v3,
                                              nullptr, OUT, 256, 512);
}
